// Attention_40037685133427
// MI455X (gfx1250) — compile-verified, matrix operand order checked by reading, not run on hardware
//
#include <hip/hip_runtime.h>
#include <hip/hip_bf16.h>

typedef _Float16 v16h __attribute__((ext_vector_type(16)));
typedef _Float16 v8h  __attribute__((ext_vector_type(8)));
typedef float    v8f  __attribute__((ext_vector_type(8)));
typedef float    v4f  __attribute__((ext_vector_type(4)));

#define B_     2
#define N_     2048
#define DIM_   1024
#define HEADS_ 16
#define DH_    64

// Load one 32-wide K-slice of an A/B operand row into the CDNA5 16-bit WMMA
// per-lane layout: lanes 0-15 take K[0..7],K[16..23]; lanes 16-31 take
// K[8..15],K[24..31].  row32 points at the first of the 32 contiguous halfs.
__device__ __forceinline__ v16h load_ab(const _Float16* row32, int lh) {
  union { v16h v; v8h h[2]; } u;
  u.h[0] = *(const v8h*)(row32 + 8 * lh);
  u.h[1] = *(const v8h*)(row32 + 16 + 8 * lh);
  return u.v;
}

__device__ __forceinline__ v8f wmma_f16(v16h a, v16h b, v8f c) {
  return __builtin_amdgcn_wmma_f32_16x16x32_f16(false, a, false, b, (short)0, c,
                                                false, false);
}

// 8 contiguous outputs per lane, vectorized.
__device__ __forceinline__ void store8(_Float16* p, v8f v) {
  v8h h;
#pragma unroll
  for (int r = 0; r < 8; ++r) h[r] = (_Float16)v[r];
  *(v8h*)p = h;
}
__device__ __forceinline__ void store8(float* p, v8f v) {
  union { v8f w; v4f q[2]; } u;
  u.w = v;
  *(v4f*)p = u.q[0];
  *(v4f*)(p + 4) = u.q[1];
}

// ---------------------------------------------------------------- FiLM cond
__global__ void __launch_bounds__(256)
cond_kernel(const float* __restrict__ emb, const float* __restrict__ W,
            const float* __restrict__ bias, float* __restrict__ cond) {
  __shared__ float s[1024];
  const int b = blockIdx.y;
  for (int i = threadIdx.x; i < 1024; i += 256) {
    float c = emb[b * 1024 + i];
    s[i] = c / (1.0f + __expf(-c));
  }
  __syncthreads();
  const int col = blockIdx.x * 256 + threadIdx.x;
  float acc = bias[col];
  for (int k = 0; k < 1024; ++k) acc += s[k] * W[k * 2048 + col];
  cond[b * 2048 + col] = acc;
}

// ------------------------------------------------------- LayerNorm + FiLM
__global__ void __launch_bounds__(256)
ln_film_kernel(const float* __restrict__ x, const float* __restrict__ gamma,
               const float* __restrict__ cond, _Float16* __restrict__ xh) {
  const int row = blockIdx.x;  // [0, B*N)
  const int b = row >> 11;     // N_ == 2048
  const float* xr = x + (size_t)row * DIM_;
  const int tid = threadIdx.x, wave = tid >> 5, lane = tid & 31;

  float v[4], sum = 0.f, sq = 0.f;
#pragma unroll
  for (int i = 0; i < 4; ++i) {
    v[i] = xr[tid + 256 * i];
    sum += v[i];
    sq += v[i] * v[i];
  }
#pragma unroll
  for (int m = 1; m <= 16; m <<= 1) {
    sum += __shfl_xor(sum, m, 32);
    sq += __shfl_xor(sq, m, 32);
  }
  __shared__ float ssum[8], ssq[8];
  if (lane == 0) { ssum[wave] = sum; ssq[wave] = sq; }
  __syncthreads();
  sum = 0.f; sq = 0.f;
#pragma unroll
  for (int w = 0; w < 8; ++w) { sum += ssum[w]; sq += ssq[w]; }
  const float mu = sum * (1.0f / DIM_);
  const float var = sq * (1.0f / DIM_) - mu * mu;
  const float rinv = rsqrtf(var + 1e-5f);

#pragma unroll
  for (int i = 0; i < 4; ++i) {
    const int c = tid + 256 * i;
    const float sc = cond[b * 2048 + c];
    const float sh = cond[b * 2048 + 1024 + c];
    float y = (v[i] - mu) * rinv * gamma[c];
    y = y * (sc + 1.0f) + sh;
    xh[(size_t)row * DIM_ + c] = (_Float16)y;
  }
}

// ---------------------------------------- weight convert+transpose (f32->f16)
__global__ void __launch_bounds__(256)
wcast_t(const float* __restrict__ W, _Float16* __restrict__ Wt, int K, int Nc) {
  const int idx = blockIdx.x * 256 + threadIdx.x;
  const int k = idx / Nc, n = idx % Nc;
  Wt[(size_t)n * K + k] = (_Float16)W[idx];
}

// ----------------------------------------------------------- WMMA GEMM
// C[token][col] = Acts[token,:] . Wt[col,:]   (both f16, K-contiguous rows)
// Computed transposed: A-operand = 16 weight columns, B-operand = 64 tokens.
// D-layout gives each lane one token row with 8 contiguous output columns per
// accumulator -> vectorized stores.
template <typename OUT>
__global__ void __launch_bounds__(256)
gemm16(const _Float16* __restrict__ Acts, const _Float16* __restrict__ Wt,
       OUT* __restrict__ C, int K, int ldc) {
  const int tid = threadIdx.x;
  const int wave = tid >> 5, lane = tid & 31;
  const int lh = lane >> 4, lm = lane & 15;
  const int cbase = (blockIdx.y * 8 + wave) * 16;  // output-column tile
  const int tbase = blockIdx.x * 64;               // token tile

  const _Float16* wrow = Wt + (size_t)(cbase + lm) * K;     // A rows
  const _Float16* arow0 = Acts + (size_t)(tbase + lm) * K;  // B rows
  const _Float16* arow1 = arow0 + (size_t)16 * K;
  const _Float16* arow2 = arow0 + (size_t)32 * K;
  const _Float16* arow3 = arow0 + (size_t)48 * K;

  v8f acc0 = {}, acc1 = {}, acc2 = {}, acc3 = {};
  for (int k0 = 0; k0 < K; k0 += 32) {
    v16h wa = load_ab(wrow + k0, lh);
    v16h b0 = load_ab(arow0 + k0, lh);
    v16h b1 = load_ab(arow1 + k0, lh);
    v16h b2 = load_ab(arow2 + k0, lh);
    v16h b3 = load_ab(arow3 + k0, lh);
    acc0 = wmma_f16(wa, b0, acc0);
    acc1 = wmma_f16(wa, b1, acc1);
    acc2 = wmma_f16(wa, b2, acc2);
    acc3 = wmma_f16(wa, b3, acc3);
  }
  const size_t col = (size_t)cbase + 8 * lh;
  store8(C + (size_t)(tbase + lm) * ldc + col, acc0);
  store8(C + (size_t)(tbase + 16 + lm) * ldc + col, acc1);
  store8(C + (size_t)(tbase + 32 + lm) * ldc + col, acc2);
  store8(C + (size_t)(tbase + 48 + lm) * ldc + col, acc3);
}

// -------------------------------------------------------- Flash attention
// Transposed formulation: S^T = K (Q*scale)^T, O^T = V^T P^T.
// Each lane owns ONE query (n = lane%16); the 32 key scores of a step live in
// its own 16 registers plus the xor-16 partner lane -> one cross-lane
// exchange per reduction, and P^T is already in B-operand layout.
// V tiles are double-buffered in LDS; staging is branchless (one b128 global
// load + 8 b16 LDS scatter stores per thread) so each key-step stays a single
// basic block and the scheduler can clause the K loads ahead of the WMMAs.
__global__ void __launch_bounds__(256)
attn_kernel(const _Float16* __restrict__ q, const _Float16* __restrict__ kv,
            _Float16* __restrict__ attn_out) {
  __shared__ __align__(16) _Float16 smemV[2][64 * 40];  // Vt[hd][key], padded
  const int tid = threadIdx.x;
  const int wave = tid >> 5, lane = tid & 31;
  const int lh = lane >> 4, lm = lane & 15;
  const int h = blockIdx.y, b = blockIdx.z;
  const int qbase = blockIdx.x * 128 + wave * 16;

  const size_t kvld = 2 * DIM_;
  const _Float16* kbase = kv + (size_t)b * N_ * kvld + h * DH_;
  const _Float16* vbase = kbase + DIM_;

  // Branchless transposed staging of V tile [jb, jb+32): thread -> one 16-B
  // chunk of one V row, scattered into Vt[hd][key].  For jb == N_ the read
  // runs a little past the KV region but stays inside our own workspace.
  const int skey = tid >> 3;        // 0..31
  const int shd = (tid & 7) * 8;    // 0,8,...,56
  auto stageV = [&](int jb, _Float16* dst) {
    const v8h chunk =
        *(const v8h*)(vbase + (size_t)(jb + skey) * kvld + shd);
#pragma unroll
    for (int j = 0; j < 8; ++j) dst[(shd + j) * 40 + skey] = chunk[j];
  };

  // Q^T B-operand with Dh^-0.5 folded in (0.125 is exact in f16).
  const _Float16* qrow = q + ((size_t)(b * N_ + qbase + lm)) * DIM_ + h * DH_;
  v16h qb0 = load_ab(qrow, lh);
  v16h qb1 = load_ab(qrow + 32, lh);
#pragma unroll
  for (int j = 0; j < 16; ++j) {
    qb0[j] *= (_Float16)0.125f;
    qb1[j] *= (_Float16)0.125f;
  }

  float m_run = -3.0e38f, l_run = 0.0f;
  v8f o0 = {}, o1 = {}, o2 = {}, o3 = {};  // O^T tiles (hd x query)

  stageV(0, smemV[0]);
  __syncthreads();

  for (int jb = 0; jb < N_; jb += 32) {
    const int cur = (jb >> 5) & 1;

    // Get the critical-path K loads in flight first.
    const _Float16* kr0 = kbase + (size_t)(jb + lm) * kvld;
    const _Float16* kr1 = kbase + (size_t)(jb + 16 + lm) * kvld;
    const v16h ka00 = load_ab(kr0, lh);
    const v16h ka01 = load_ab(kr0 + 32, lh);
    const v16h ka10 = load_ab(kr1, lh);
    const v16h ka11 = load_ab(kr1 + 32, lh);

    // Stage the next V tile into the other buffer (overlaps the WMMAs).
    stageV(jb + 32, smemV[1 - cur]);
    __builtin_prefetch(kbase + (size_t)(jb + 32) * kvld, 0, 1);

    v8f s0 = {}, s1 = {};
    s0 = wmma_f16(ka00, qb0, s0);
    s1 = wmma_f16(ka10, qb0, s1);
    s0 = wmma_f16(ka01, qb1, s0);
    s1 = wmma_f16(ka11, qb1, s1);

    // Online softmax over this lane's query (scores already scaled).
    float mx = fmaxf(s0[0], s1[0]);
#pragma unroll
    for (int j = 1; j < 8; ++j) mx = fmaxf(mx, fmaxf(s0[j], s1[j]));
    mx = fmaxf(mx, __shfl_xor(mx, 16, 32));  // other half of the 32 keys
    const float mnew = fmaxf(m_run, mx);

    union { v16h v; _Float16 e[16]; } pb;  // P^T B-operand, in-lane pack
    float sv = 0.f;
#pragma unroll
    for (int j = 0; j < 8; ++j) {
      const float p0 = __expf(s0[j] - mnew);
      const float p1 = __expf(s1[j] - mnew);
      sv += p0 + p1;
      pb.e[j] = (_Float16)p0;
      pb.e[8 + j] = (_Float16)p1;
    }
    sv += __shfl_xor(sv, 16, 32);
    const float alpha = __expf(m_run - mnew);
    l_run = l_run * alpha + sv;
    m_run = mnew;
#pragma unroll
    for (int r = 0; r < 8; ++r) {
      o0[r] *= alpha; o1[r] *= alpha; o2[r] *= alpha; o3[r] *= alpha;
    }

    // O^T += V^T * P^T; A-operand rows = Vt rows (contiguous in LDS).
    const _Float16* vt = smemV[cur];
    const v16h va0 = load_ab(vt + (lm) * 40, lh);
    const v16h va1 = load_ab(vt + (16 + lm) * 40, lh);
    const v16h va2 = load_ab(vt + (32 + lm) * 40, lh);
    const v16h va3 = load_ab(vt + (48 + lm) * 40, lh);
    o0 = wmma_f16(va0, pb.v, o0);
    o1 = wmma_f16(va1, pb.v, o1);
    o2 = wmma_f16(va2, pb.v, o2);
    o3 = wmma_f16(va3, pb.v, o3);

    __syncthreads();  // staging of next tile complete / current reads done
  }

  // Lane owns token row (qbase+lm); hd chunks are contiguous -> b128 stores.
  const float inv = 1.0f / l_run;
  _Float16* orow = attn_out + ((size_t)(b * N_ + qbase + lm)) * DIM_ + h * DH_;
  v8f w0 = o0 * inv, w1 = o1 * inv, w2 = o2 * inv, w3 = o3 * inv;
  store8(orow + 8 * lh, w0);
  store8(orow + 16 + 8 * lh, w1);
  store8(orow + 32 + 8 * lh, w2);
  store8(orow + 48 + 8 * lh, w3);
}

// ------------------------------------------------------------------ launch
extern "C" void kernel_launch(void* const* d_in, const int* in_sizes, int n_in,
                              void* d_out, int out_size, void* d_ws,
                              size_t ws_size, hipStream_t stream) {
  (void)in_sizes; (void)n_in; (void)out_size; (void)ws_size;
  const float* x      = (const float*)d_in[0];
  const float* cemb   = (const float*)d_in[1];
  const float* gamma  = (const float*)d_in[2];
  const float* cond_W = (const float*)d_in[3];
  const float* cond_b = (const float*)d_in[4];
  const float* Wq     = (const float*)d_in[5];
  const float* Wkv    = (const float*)d_in[6];
  const float* Wo     = (const float*)d_in[7];
  float* out = (float*)d_out;

  char* ws = (char*)d_ws;
  const size_t MB = 1024 * 1024;
  _Float16* xh   = (_Float16*)(ws);            //  8 MB
  _Float16* qb   = (_Float16*)(ws + 8 * MB);   //  8 MB
  _Float16* kvb  = (_Float16*)(ws + 16 * MB);  // 16 MB
  _Float16* attn = (_Float16*)(ws + 32 * MB);  //  8 MB (also tail-read pad)
  _Float16* WqT  = (_Float16*)(ws + 40 * MB);  //  2 MB
  _Float16* WkvT = (_Float16*)(ws + 42 * MB);  //  4 MB
  _Float16* WoT  = (_Float16*)(ws + 46 * MB);  //  2 MB
  float*    cond = (float*)(ws + 48 * MB);     // 16 KB

  cond_kernel<<<dim3(2048 / 256, B_), 256, 0, stream>>>(cemb, cond_W, cond_b,
                                                        cond);
  ln_film_kernel<<<B_ * N_, 256, 0, stream>>>(x, gamma, cond, xh);
  wcast_t<<<(1024 * 1024) / 256, 256, 0, stream>>>(Wq, WqT, 1024, 1024);
  wcast_t<<<(1024 * 2048) / 256, 256, 0, stream>>>(Wkv, WkvT, 1024, 2048);
  wcast_t<<<(1024 * 1024) / 256, 256, 0, stream>>>(Wo, WoT, 1024, 1024);

  // grid: x = token tiles (M/64), y = column tiles (Nc/128)
  gemm16<_Float16><<<dim3((B_ * N_) / 64, 1024 / 128), 256, 0, stream>>>(
      xh, WqT, qb, 1024, 1024);
  gemm16<_Float16><<<dim3((B_ * N_) / 64, 2048 / 128), 256, 0, stream>>>(
      xh, WkvT, kvb, 1024, 2048);

  attn_kernel<<<dim3(N_ / 128, HEADS_, B_), 256, 0, stream>>>(qb, kvb, attn);

  gemm16<float><<<dim3((B_ * N_) / 64, 1024 / 128), 256, 0, stream>>>(
      attn, WoT, out, 1024, 1024);
}
